// SelfAttention_3599182594771
// MI455X (gfx1250) — matrix operand order repaired and checked by reading, not compiled, not run on hardware
//
#include <hip/hip_runtime.h>
#include <hip/hip_bf16.h>

typedef unsigned short u16;
typedef __attribute__((ext_vector_type(16))) __bf16 v16bf;
typedef __attribute__((ext_vector_type(8)))  float  v8f;

#define HIDDEN 1024
#define SEQ    2048
#define BATCH  4
#define NH     16
#define HD     64
#define ROWS   (BATCH * SEQ)   // 8192

// fp32 -> bf16 round-to-nearest-even
__device__ __forceinline__ u16 f2bf(float f) {
    unsigned int u = __float_as_uint(f);
    return (u16)((u + 0x7FFFu + ((u >> 16) & 1u)) >> 16);
}

// 16 bf16 fragment (32 bytes) with uint4 access for vector LDS/global loads
union Frag {
    v16bf v;
    uint4 q[2];
};

// ---------------------------------------------------------------------------
// CDNA5 async global->LDS staging (ASYNCcnt path, ISA 15.18.3 op 98).
// VDST VGPR holds the workgroup-relative LDS byte offset = low 32 bits of a
// flat pointer to a __shared__ object.
// ---------------------------------------------------------------------------
__device__ __forceinline__ void async_b128(const u16* g, u16* lds) {
    unsigned off = (unsigned)(unsigned long long)(const void*)lds;
    asm volatile("global_load_async_to_lds_b128 %0, %1, off"
                 :: "v"(off), "v"(g) : "memory");
}
__device__ __forceinline__ void wait_async() {
    asm volatile("s_wait_asynccnt 0" ::: "memory");
}
// wave-private LDS ordering (LDS is in-order within a wave; this pins the
// compiler ordering for the cross-lane P-tile exchange without a block barrier)
__device__ __forceinline__ void wave_fence() {
    __builtin_amdgcn_fence(__ATOMIC_ACQ_REL, "wavefront");
}

// ---------------------------------------------------------------------------
// Kernel 1: LayerNorm row -> bf16 h
// ---------------------------------------------------------------------------
__global__ __launch_bounds__(256) void ln_bf16_kernel(
    const float* __restrict__ x, const float* __restrict__ g,
    const float* __restrict__ b, u16* __restrict__ h) {
    int row = blockIdx.x;
    int t   = threadIdx.x;
    const float* xr = x + (size_t)row * HIDDEN;
    float4 v = *(const float4*)(xr + t * 4);
    __shared__ float r1[256], r2[256];
    r1[t] = v.x + v.y + v.z + v.w;
    r2[t] = v.x * v.x + v.y * v.y + v.z * v.z + v.w * v.w;
    __syncthreads();
    for (int o = 128; o > 0; o >>= 1) {
        if (t < o) { r1[t] += r1[t + o]; r2[t] += r2[t + o]; }
        __syncthreads();
    }
    float mu   = r1[0] * (1.0f / HIDDEN);
    float var  = r2[0] * (1.0f / HIDDEN) - mu * mu;
    float rstd = rsqrtf(var + 1e-5f);
    float4 gv = *(const float4*)(g + t * 4);
    float4 bv = *(const float4*)(b + t * 4);
    u16* out = h + (size_t)row * HIDDEN + t * 4;
    out[0] = f2bf((v.x - mu) * rstd * gv.x + bv.x);
    out[1] = f2bf((v.y - mu) * rstd * gv.y + bv.y);
    out[2] = f2bf((v.z - mu) * rstd * gv.z + bv.z);
    out[3] = f2bf((v.w - mu) * rstd * gv.w + bv.w);
}

// ---------------------------------------------------------------------------
// Kernel 2: convert the four 1024x1024 fp32 weight matrices to bf16 (packed)
// ---------------------------------------------------------------------------
__global__ __launch_bounds__(256) void wcvt_kernel(
    const float* __restrict__ W0, const float* __restrict__ W1,
    const float* __restrict__ W2, const float* __restrict__ W3,
    u16* __restrict__ dst) {
    size_t i = (size_t)blockIdx.x * blockDim.x + threadIdx.x;  // 4M total
    int j = (int)(i >> 20);
    const float* W = (j == 0) ? W0 : (j == 1) ? W1 : (j == 2) ? W2 : W3;
    dst[i] = f2bf(W[i & 0xFFFFFu]);
}

// ---------------------------------------------------------------------------
// Shared GEMM body: wave tile 32x64 over block tile 128x128, K-step 32.
// 8 waves = 4(M) x 2(N). Double-buffered LDS; async DMA of tile k+1 overlaps
// WMMA on tile k; a single block barrier per K-step.
// ---------------------------------------------------------------------------
struct GemmCore {
    v8f acc[2][4];
    int mBase, nBase, t, w, waveM, waveN, lhalf, ln;
};

__device__ __forceinline__ void gemm_init(GemmCore& gc) {
    gc.mBase = blockIdx.x * 128;
    gc.nBase = blockIdx.y * 128;
    gc.t = threadIdx.x;
    int l = gc.t & 31;
    gc.w = gc.t >> 5;
    gc.waveM = gc.w & 3;
    gc.waveN = gc.w >> 2;
    gc.lhalf = (l < 16) ? 0 : 1;
    gc.ln = l & 15;
}

__device__ __forceinline__ void gemm_stage(
    const GemmCore& gc, const u16* __restrict__ A, const u16* __restrict__ W,
    u16* sA, u16* sW, int k0) {
    // each tile is 128 rows x 32 k: 512 16B chunks, 2 per thread per tile
    for (int c = 0; c < 2; c++) {
        int ch = gc.t * 2 + c;
        int r = ch >> 2, kc = (ch & 3) * 8;
        async_b128(A + (size_t)(gc.mBase + r) * HIDDEN + k0 + kc, sA + r * 32 + kc);
        async_b128(W + (size_t)(gc.nBase + r) * HIDDEN + k0 + kc, sW + r * 32 + kc);
    }
}

__device__ __forceinline__ void gemm_core(
    GemmCore& gc, const u16* __restrict__ A, const u16* __restrict__ W,
    u16 (*sA)[128 * 32], u16 (*sW)[128 * 32]) {
    for (int fm = 0; fm < 2; fm++)
        for (int fn = 0; fn < 4; fn++) gc.acc[fm][fn] = (v8f){};

    gemm_stage(gc, A, W, sA[0], sW[0], 0);            // prologue: fill buf 0
    for (int k0 = 0; k0 < HIDDEN; k0 += 32) {
        int cur = (k0 >> 5) & 1;
        wait_async();          // buf[cur] landed (issued one iteration ago)
        __syncthreads();       // all waves' DMA visible; all done reading buf[cur^1]
        if (k0 + 32 < HIDDEN)  // overlap DMA of next tile with this tile's math
            gemm_stage(gc, A, W, sA[cur ^ 1], sW[cur ^ 1], k0 + 32);

        Frag a[2], b[4];
        for (int fm = 0; fm < 2; fm++) {           // A 16x32 layout
            const u16* p = sA[cur] + (gc.waveM * 32 + fm * 16 + gc.ln) * 32;
            a[fm].q[0] = *(const uint4*)(p + gc.lhalf * 8);
            a[fm].q[1] = *(const uint4*)(p + 16 + gc.lhalf * 8);
        }
        for (int fn = 0; fn < 4; fn++) {           // B 32x16 layout (table K order)
            const u16* p = sW[cur] + (gc.waveN * 64 + fn * 16 + gc.ln) * 32;
            b[fn].q[0] = *(const uint4*)(p + gc.lhalf * 8);
            b[fn].q[1] = *(const uint4*)(p + 16 + gc.lhalf * 8);
        }
        for (int fm = 0; fm < 2; fm++)
            for (int fn = 0; fn < 4; fn++)
                gc.acc[fm][fn] = __builtin_amdgcn_wmma_f32_16x16x32_bf16(
                    false, a[fm].v, false, b[fn].v, (short)0, gc.acc[fm][fn],
                    false, false);
    }
}

// ---------------------------------------------------------------------------
// Kernel 3: fused Q/K/V projection GEMM  y = h @ W^T + bias  (bf16 out)
//   grid.z: 0=Q (scaled by 0.125), 1=K, 2=V
// ---------------------------------------------------------------------------
__global__ __launch_bounds__(256) void gemm_qkv_kernel(
    const u16* __restrict__ A, const u16* __restrict__ Wbf,
    const float* __restrict__ bq, const float* __restrict__ bk,
    const float* __restrict__ bv,
    u16* __restrict__ qb, u16* __restrict__ kb, u16* __restrict__ vb) {
    int z = blockIdx.z;
    const u16*   W    = Wbf + (size_t)z * HIDDEN * HIDDEN;
    const float* bias = (z == 0) ? bq : (z == 1) ? bk : bv;
    u16*         out  = (z == 0) ? qb : (z == 1) ? kb : vb;
    float scale = (z == 0) ? 0.125f : 1.0f;  // fold 1/sqrt(64) into Q

    __shared__ u16 sA[2][128 * 32];
    __shared__ u16 sW[2][128 * 32];
    GemmCore gc;
    gemm_init(gc);
    gemm_core(gc, A, W, sA, sW);

    for (int fm = 0; fm < 2; fm++)
        for (int fn = 0; fn < 4; fn++) {
            int col = gc.nBase + gc.waveN * 64 + fn * 16 + gc.ln;
            float bc = bias[col];
            for (int i = 0; i < 8; i++) {
                int row = gc.mBase + gc.waveM * 32 + fm * 16 + gc.lhalf * 8 + i;
                out[(size_t)row * HIDDEN + col] = f2bf((gc.acc[fm][fn][i] + bc) * scale);
            }
        }
}

// ---------------------------------------------------------------------------
// Kernel 5: output projection  out = ctx @ Wo^T + bo + x   (fp32 out)
// ---------------------------------------------------------------------------
__global__ __launch_bounds__(256) void gemm_out_kernel(
    const u16* __restrict__ A, const u16* __restrict__ W,
    const float* __restrict__ bo, const float* __restrict__ x,
    float* __restrict__ out) {
    __shared__ u16 sA[2][128 * 32];
    __shared__ u16 sW[2][128 * 32];
    GemmCore gc;
    gemm_init(gc);
    gemm_core(gc, A, W, sA, sW);

    for (int fm = 0; fm < 2; fm++)
        for (int fn = 0; fn < 4; fn++) {
            int col = gc.nBase + gc.waveN * 64 + fn * 16 + gc.ln;
            float bc = bo[col];
            for (int i = 0; i < 8; i++) {
                int row = gc.mBase + gc.waveM * 32 + fm * 16 + gc.lhalf * 8 + i;
                size_t idx = (size_t)row * HIDDEN + col;
                out[idx] = gc.acc[fm][fn][i] + bc + x[idx];
            }
        }
}

// ---------------------------------------------------------------------------
// Kernel 4: flash attention per (64-row Q tile, head).
//   128 threads = 4 waves; each wave owns 16 Q rows.
//   K tiles async-DMA'd, V transposed on store, both double-buffered; one
//   block barrier per tile. P goes through wave-private LDS (C->A layout)
//   guarded by a wavefront fence only.
// ---------------------------------------------------------------------------
__global__ __launch_bounds__(128) void attn_kernel(
    const u16* __restrict__ qbuf, const u16* __restrict__ kbuf,
    const u16* __restrict__ vbuf, u16* __restrict__ ctx) {
    int qTile = blockIdx.x;           // 0..31
    int bh    = blockIdx.y;           // 0..63
    int bIdx  = bh >> 4, hIdx = bh & 15;
    int t = threadIdx.x, l = t & 31, w = t >> 5;
    int lhalf = (l < 16) ? 0 : 1, ln = l & 15;
    size_t rowBase = (size_t)bIdx * SEQ;
    int qBase = qTile * 64, colBase = hIdx * HD;

    __shared__ u16 sK[2][64 * 64];     // (key, d) row-major
    __shared__ u16 sVt[2][64 * 64];    // (d, key) row-major (transposed V)
    __shared__ u16 sP[4][16 * 64];     // per-wave P tile

    Frag qa[2];
    {   // Q A-frags, Q already carries the 1/sqrt(d) scale
        const u16* p = qbuf + (rowBase + qBase + w * 16 + ln) * HIDDEN + colBase;
        for (int ks = 0; ks < 2; ks++) {
            qa[ks].q[0] = *(const uint4*)(p + ks * 32 + lhalf * 8);
            qa[ks].q[1] = *(const uint4*)(p + ks * 32 + 16 + lhalf * 8);
        }
    }

    v8f acc[4] = {};
    float mrow[8], lrow[8];
    for (int i = 0; i < 8; i++) { mrow[i] = -3.0e38f; lrow[i] = 0.0f; }

    // tile stager: async K rows + scalar transpose of V
    auto stage = [&](int buf, int kBase) {
        for (int c = 0; c < 4; c++) {              // K: 512 16B chunks / 128 thr
            int ch = t * 4 + c;
            int key = ch >> 3, ds = (ch & 7) * 8;
            async_b128(kbuf + (rowBase + kBase + key) * HIDDEN + colBase + ds,
                       sK[buf] + key * 64 + ds);
        }
        for (int i = 0; i < 32; i++) {             // V, transposed on store
            int e = t + i * 128;
            int key = e >> 6, d = e & 63;
            sVt[buf][d * 64 + key] = vbuf[(rowBase + kBase + key) * HIDDEN + colBase + d];
        }
    };

    stage(0, 0);   // prologue
    for (int kt = 0; kt < SEQ / 64; kt++) {
        int cur = kt & 1;
        wait_async();          // sK[cur] DMA landed
        __syncthreads();       // all waves' stores visible; buf[cur^1] reads done
        if (kt + 1 < SEQ / 64) stage(cur ^ 1, (kt + 1) * 64);

        // scores S = Q . K^T  (16x64 per wave)
        v8f s[4] = {};
        for (int nf = 0; nf < 4; nf++)
            for (int ks = 0; ks < 2; ks++) {
                Frag kb;
                const u16* p = sK[cur] + (nf * 16 + ln) * 64 + ks * 32;
                kb.q[0] = *(const uint4*)(p + lhalf * 8);
                kb.q[1] = *(const uint4*)(p + 16 + lhalf * 8);
                s[nf] = __builtin_amdgcn_wmma_f32_16x16x32_bf16(
                    false, qa[ks].v, false, kb.v, (short)0, s[nf], false, false);
            }

        // online softmax: row lives on 16 lanes of one half; xor 1/2/4/8
        for (int i = 0; i < 8; i++) {
            float rm = fmaxf(fmaxf(s[0][i], s[1][i]), fmaxf(s[2][i], s[3][i]));
            rm = fmaxf(rm, __shfl_xor(rm, 1, 32));
            rm = fmaxf(rm, __shfl_xor(rm, 2, 32));
            rm = fmaxf(rm, __shfl_xor(rm, 4, 32));
            rm = fmaxf(rm, __shfl_xor(rm, 8, 32));
            float mnew = fmaxf(mrow[i], rm);
            float corr = __expf(mrow[i] - mnew);
            float rs = 0.0f;
            for (int nf = 0; nf < 4; nf++) {
                float p = __expf(s[nf][i] - mnew);
                s[nf][i] = p;
                rs += p;
            }
            rs += __shfl_xor(rs, 1, 32);
            rs += __shfl_xor(rs, 2, 32);
            rs += __shfl_xor(rs, 4, 32);
            rs += __shfl_xor(rs, 8, 32);
            lrow[i] = lrow[i] * corr + rs;
            mrow[i] = mnew;
            for (int nf = 0; nf < 4; nf++) acc[nf][i] *= corr;
        }

        // C-layout -> A-layout for P via wave-private LDS (no block barrier)
        for (int nf = 0; nf < 4; nf++)
            for (int i = 0; i < 8; i++)
                sP[w][(lhalf * 8 + i) * 64 + nf * 16 + ln] = f2bf(s[nf][i]);
        wave_fence();

        Frag pa[2];
        for (int ks = 0; ks < 2; ks++) {
            const u16* pp = sP[w] + ln * 64;
            pa[ks].q[0] = *(const uint4*)(pp + ks * 32 + lhalf * 8);
            pa[ks].q[1] = *(const uint4*)(pp + ks * 32 + 16 + lhalf * 8);
        }
        for (int nf = 0; nf < 4; nf++)
            for (int ks = 0; ks < 2; ks++) {
                Frag vbf;
                const u16* vp = sVt[cur] + (nf * 16 + ln) * 64 + ks * 32;
                vbf.q[0] = *(const uint4*)(vp + lhalf * 8);
                vbf.q[1] = *(const uint4*)(vp + 16 + lhalf * 8);
                acc[nf] = __builtin_amdgcn_wmma_f32_16x16x32_bf16(
                    false, pa[ks].v, false, vbf.v, (short)0, acc[nf], false, false);
            }
    }

    for (int nf = 0; nf < 4; nf++) {
        int col = colBase + nf * 16 + ln;
        for (int i = 0; i < 8; i++) {
            float inv = 1.0f / lrow[i];
            int qrow = qBase + w * 16 + lhalf * 8 + i;
            ctx[(rowBase + qrow) * HIDDEN + col] = f2bf(acc[nf][i] * inv);
        }
    }
}

// ---------------------------------------------------------------------------
extern "C" void kernel_launch(void* const* d_in, const int* in_sizes, int n_in,
                              void* d_out, int out_size, void* d_ws, size_t ws_size,
                              hipStream_t stream) {
    const float* x  = (const float*)d_in[0];
    const float* Wq = (const float*)d_in[1];
    const float* bq = (const float*)d_in[2];
    const float* Wk = (const float*)d_in[3];
    const float* bk = (const float*)d_in[4];
    const float* Wv = (const float*)d_in[5];
    const float* bv = (const float*)d_in[6];
    const float* Wo = (const float*)d_in[7];
    const float* bo = (const float*)d_in[8];
    const float* g  = (const float*)d_in[9];
    const float* be = (const float*)d_in[10];
    float* out = (float*)d_out;

    const size_t MB = 1024ull * 1024ull;
    char* ws  = (char*)d_ws;
    u16* h    = (u16*)(ws);               // 16 MB: LN'd activations, bf16
    u16* wbf  = (u16*)(ws + 16 * MB);     //  8 MB: Wq,Wk,Wv,Wo bf16
    u16* qb   = (u16*)(ws + 24 * MB);     // 16 MB
    u16* kb   = (u16*)(ws + 40 * MB);     // 16 MB
    u16* vb   = (u16*)(ws + 56 * MB);     // 16 MB
    u16* ctx  = (u16*)(ws + 72 * MB);     // 16 MB

    ln_bf16_kernel<<<ROWS, 256, 0, stream>>>(x, g, be, h);
    wcvt_kernel<<<16384, 256, 0, stream>>>(Wq, Wk, Wv, Wo, wbf);
    gemm_qkv_kernel<<<dim3(ROWS / 128, HIDDEN / 128, 3), 256, 0, stream>>>(
        h, wbf, bq, bk, bv, qb, kb, vb);
    attn_kernel<<<dim3(SEQ / 64, BATCH * NH), 128, 0, stream>>>(qb, kb, vb, ctx);
    gemm_out_kernel<<<dim3(ROWS / 128, HIDDEN / 128), 256, 0, stream>>>(
        ctx, wbf + 3 * MB, bo, x, out);
}
